// MultiTaskTrunkNetwork_61177514164384
// MI455X (gfx1250) — matrix operand order repaired and checked by reading, not compiled, not run on hardware
//
#include <hip/hip_runtime.h>

#define B_TOTAL 262144
#define IN_DIM  89
#define H_DIM   64
#define OUT_DIM 8
#define T_TASKS 50

typedef __bf16 bf16;
typedef __attribute__((ext_vector_type(16))) __bf16 v16bf;
typedef __attribute__((ext_vector_type(8)))  float  v8f;

__device__ __forceinline__ float fast_tanh(float x) {
    // tanh(x) = 1 - 2/(exp(2x)+1); branchless, saturates to +/-1 via inf/0.
    float e = __builtin_amdgcn_exp2f(x * 2.885390081777927f);   // exp(2x)
    float r = __builtin_amdgcn_rcpf(e + 1.0f);
    return fmaf(-2.0f, r, 1.0f);
}

// A-fragment element index -> K offset within a 32-wide k-step
// (16-bit A 16x32 layout: lanes 0-15 K=0..7/16..23, lanes 16-31 K=8..15/24..31)
__device__ __forceinline__ int a_elem_k(int halfA, int i) {
    return (i < 8) ? (halfA * 8 + i) : (16 + halfA * 8 + (i - 8));
}

// Store h(row r, col c) into A-fragment-ordered staging: slot (s, lane)=(c>>5, r+16*halfA)
__device__ __forceinline__ void store_h(bf16* hbase, int r, int c, float val) {
    int s     = c >> 5;
    int w     = c & 31;
    int halfA = (w & 15) >> 3;
    int i     = (w & 7) + ((w & 16) ? 8 : 0);
    hbase[((s * 32) + (r + 16 * halfA)) * 16 + i] = (bf16)val;
}

__global__ __launch_bounds__(256) void mtl_fused(
    const float* __restrict__ X,  const int* __restrict__ tasks,
    const float* __restrict__ W0, const float* __restrict__ b0,
    const float* __restrict__ W1, const float* __restrict__ b1,
    const float* __restrict__ W2, const float* __restrict__ b2,
    const float* __restrict__ hW, const float* __restrict__ hB,
    float* __restrict__ out)
{
    // B-fragment-contiguous weight storage: [frag][lane][16 bf16]
    __shared__ __align__(32) bf16 sW0[12 * 32 * 16];  // 12 KB (K padded 89->96)
    __shared__ __align__(32) bf16 sW1[ 8 * 32 * 16];  //  8 KB
    __shared__ __align__(32) bf16 sW2[ 8 * 32 * 16];  //  8 KB
    __shared__ float sB[3][H_DIM];
    __shared__ float sHB[T_TASKS * OUT_DIM];
    // h staging in A-fragment order: [wave][s(2)][lane(32)][16 bf16] = 16 KB
    __shared__ __align__(32) bf16 hstage[8][2 * 32 * 16];

    const int tid = threadIdx.x;

    // ---- cooperative weight staging (fp32 global -> bf16 B-fragment layout) ----
    // frag f = s*4+t ; lane l -> n = t*16 + (l&15), kh = l>>4 ; elem i -> k = s*32 + a_elem_k(kh, i)
    for (int idx = tid; idx < 12 * 512; idx += 256) {
        int f = idx >> 9, r = idx & 511, l = r >> 4, i = r & 15;
        int s = f >> 2,  t = f & 3;
        int n = t * 16 + (l & 15);
        int k = s * 32 + a_elem_k(l >> 4, i);
        sW0[idx] = (bf16)((k < IN_DIM) ? W0[k * H_DIM + n] : 0.f);
    }
    for (int idx = tid; idx < 8 * 512; idx += 256) {
        int f = idx >> 9, r = idx & 511, l = r >> 4, i = r & 15;
        int s = f >> 2,  t = f & 3;
        int n = t * 16 + (l & 15);
        int k = s * 32 + a_elem_k(l >> 4, i);
        sW1[idx] = (bf16)(W1[k * H_DIM + n]);
        sW2[idx] = (bf16)(W2[k * H_DIM + n]);
    }
    if (tid < H_DIM) { sB[0][tid] = b0[tid]; sB[1][tid] = b1[tid]; sB[2][tid] = b2[tid]; }
    for (int idx = tid; idx < T_TASKS * OUT_DIM; idx += 256) sHB[idx] = hB[idx];
    __syncthreads();

    const int wave = tid >> 5;
    const int lane = tid & 31;
    const int m    = lane & 15;
    const int half = lane >> 4;
    const size_t rowBase = (size_t)blockIdx.x * 128 + (size_t)wave * 16;
    const size_t row = rowBase + m;

    bf16* hbase = hstage[wave];
    const v16bf* w0f = (const v16bf*)sW0;

    // ---- layer 0: A fragments straight from global X, K zero-padded to 96 ----
    v16bf a0[3];
    const float* xrow = X + row * IN_DIM;   // rows only 4B aligned -> scalar loads
#pragma unroll
    for (int s = 0; s < 3; ++s) {
        v16bf a;
#pragma unroll
        for (int i = 0; i < 16; ++i) {
            int k = s * 32 + a_elem_k(half, i);
            a[i] = (bf16)((k < IN_DIM) ? xrow[k] : 0.f);
        }
        a0[s] = a;
    }

#pragma unroll
    for (int t = 0; t < 4; ++t) {
        v8f c = {0.f, 0.f, 0.f, 0.f, 0.f, 0.f, 0.f, 0.f};
#pragma unroll
        for (int s = 0; s < 3; ++s) {
            c = __builtin_amdgcn_wmma_f32_16x16x32_bf16(
                    false, a0[s], false, w0f[(s * 4 + t) * 32 + lane],
                    (short)0, c, false, false);
        }
        float bias = sB[0][t * 16 + m];          // C layout: col n = lane&15
#pragma unroll
        for (int v = 0; v < 8; ++v) {
            store_h(hbase, v + 8 * half, t * 16 + m, fast_tanh(c[v] + bias));
        }
    }

    // ---- layers 1 & 2: contiguous A-fragment loads from wave-private LDS ----
#pragma unroll
    for (int layer = 1; layer <= 2; ++layer) {
        const v16bf* wf = (const v16bf*)(layer == 1 ? sW1 : sW2);
        const float* bv = sB[layer];
        v16bf a[2];
#pragma unroll
        for (int s = 0; s < 2; ++s)
            a[s] = ((const v16bf*)hbase)[s * 32 + lane];
#pragma unroll
        for (int t = 0; t < 4; ++t) {
            v8f c = {0.f, 0.f, 0.f, 0.f, 0.f, 0.f, 0.f, 0.f};
#pragma unroll
            for (int s = 0; s < 2; ++s) {
                c = __builtin_amdgcn_wmma_f32_16x16x32_bf16(
                        false, a[s], false, wf[(s * 4 + t) * 32 + lane],
                        (short)0, c, false, false);
            }
            float bias = bv[t * 16 + m];
#pragma unroll
            for (int v = 0; v < 8; ++v) {
                store_h(hbase, v + 8 * half, t * 16 + m, fast_tanh(c[v] + bias));
            }
        }
    }

    // ---- per-task head: lane pair splits K (32 each), coalesced float4 weight reads ----
    const int task = tasks[row];
    // sample m's k-range [half*32, half*32+32) lives in slots (s=half, lanes m and m+16)
    v16bf q0 = ((const v16bf*)hbase)[half * 32 + m];        // k_local {0..7,16..23}
    v16bf q1 = ((const v16bf*)hbase)[half * 32 + m + 16];   // k_local {8..15,24..31}

    const float4* wp = (const float4*)(hW + ((size_t)task * H_DIM + half * 32) * OUT_DIM);
    float acc[8] = {0.f, 0.f, 0.f, 0.f, 0.f, 0.f, 0.f, 0.f};
#pragma unroll
    for (int i = 0; i < 16; ++i) {
        int k0 = (i < 8) ? i : (i + 8);   // q0 element -> k_local
        int k1 = k0 + 8;                  // q1 element -> k_local
        float h0 = (float)q0[i];
        float h1 = (float)q1[i];
        float4 wa = wp[2 * k0], wb = wp[2 * k0 + 1];
        float4 wc = wp[2 * k1], wd = wp[2 * k1 + 1];
        acc[0] += h0 * wa.x + h1 * wc.x;
        acc[1] += h0 * wa.y + h1 * wc.y;
        acc[2] += h0 * wa.z + h1 * wc.z;
        acc[3] += h0 * wa.w + h1 * wc.w;
        acc[4] += h0 * wb.x + h1 * wd.x;
        acc[5] += h0 * wb.y + h1 * wd.y;
        acc[6] += h0 * wb.z + h1 * wd.z;
        acc[7] += h0 * wb.w + h1 * wd.w;
    }
#pragma unroll
    for (int j = 0; j < 8; ++j) acc[j] += __shfl_xor(acc[j], 16, 32);

    if (half == 0) {
        float4 o0, o1;
        o0.x = acc[0] + sHB[task * OUT_DIM + 0];
        o0.y = acc[1] + sHB[task * OUT_DIM + 1];
        o0.z = acc[2] + sHB[task * OUT_DIM + 2];
        o0.w = acc[3] + sHB[task * OUT_DIM + 3];
        o1.x = acc[4] + sHB[task * OUT_DIM + 4];
        o1.y = acc[5] + sHB[task * OUT_DIM + 5];
        o1.z = acc[6] + sHB[task * OUT_DIM + 6];
        o1.w = acc[7] + sHB[task * OUT_DIM + 7];
        float4* op = (float4*)(out + row * OUT_DIM);
        op[0] = o0;
        op[1] = o1;
    }
}

extern "C" void kernel_launch(void* const* d_in, const int* in_sizes, int n_in,
                              void* d_out, int out_size, void* d_ws, size_t ws_size,
                              hipStream_t stream) {
    const float* X     = (const float*)d_in[0];
    const int*   tasks = (const int*)  d_in[1];
    const float* W0    = (const float*)d_in[2];
    const float* b0    = (const float*)d_in[3];
    const float* W1    = (const float*)d_in[4];
    const float* b1    = (const float*)d_in[5];
    const float* W2    = (const float*)d_in[6];
    const float* b2    = (const float*)d_in[7];
    const float* hW    = (const float*)d_in[8];
    const float* hB    = (const float*)d_in[9];
    float* out = (float*)d_out;

    dim3 grid(B_TOTAL / 128);   // 2048 blocks x 8 waves x 16 samples
    mtl_fused<<<grid, 256, 0, stream>>>(X, tasks, W0, b0, W1, b1, W2, b2, hW, hB, out);
}
